// ScalpingMasterMCA_87359634801400
// MI455X (gfx1250) — matrix operand order repaired and checked by reading, not compiled, not run on hardware
//
#include <hip/hip_runtime.h>
#include <hip/hip_bf16.h>

// ---------------------------------------------------------------------------
// MI455X (gfx1250) implementation. All contractions run on
// v_wmma_f32_16x16x32_f16 (f16 operands, f32 accumulate). wave32 everywhere.
// x staging uses GLOBAL_LOAD_ASYNC_TO_LDS_B128 (ASYNCcnt) - CDNA5 async path.
//
// ws layout (bytes):
//   [0, B*64*4)            : long_context  (f32, B x 64)
//   [+0 .. +64)            : b_u (16 f32, padded)
//   [+64 ...)              : packed f16 B-operand tables (u32 words)
// Total ~4.2 MB of d_ws used.
// ---------------------------------------------------------------------------

typedef __attribute__((ext_vector_type(16))) _Float16 v16h;
typedef __attribute__((ext_vector_type(8)))  float    v8f;

#define H_DIM  64
#define DST    16
#define SLONG  100
#define SSHORT 10

union AOp { unsigned u[8]; v16h v; };

__device__ __forceinline__ unsigned pkh(float a, float b) {
  union { _Float16 h[2]; unsigned u; } t;
  t.h[0] = (_Float16)a; t.h[1] = (_Float16)b; return t.u;
}
__device__ __forceinline__ unsigned ldpair(const _Float16* p) {
  return *(const unsigned*)p;   // p always 4B aligned (even half index)
}
// A-operand K-slot for (vgpr r, lane-half g): contiguous k = 16g + 2r (B packs use the same order)
__device__ __forceinline__ int kpA(int r, int g) {
  return (g << 4) + (r << 1);
}
// Single-instruction transcendentals: v_tanh_f32 if the toolchain exposes it,
// otherwise v_exp_f32 + v_rcp_f32 (no IEEE div sequence on the serial path).
__device__ __forceinline__ float tanh_f(float x) {
#if __has_builtin(__builtin_amdgcn_tanhf)
  return __builtin_amdgcn_tanhf(x);
#else
  float e = __expf(2.f * x);
  return 1.f - 2.f * __builtin_amdgcn_rcpf(e + 1.f);
#endif
}
__device__ __forceinline__ float sigm_f(float x) {
  return __builtin_amdgcn_rcpf(1.f + __expf(-x));
}
__device__ __forceinline__ v8f wmma_f16(const AOp& a, const AOp& b, v8f c) {
  return __builtin_amdgcn_wmma_f32_16x16x32_f16(
      false, a.v, false, b.v, (short)0, c, false, false);
}
// CDNA5 async copy: global -> LDS, 16B per lane, tracked on ASYNCcnt.
__device__ __forceinline__ void async_load_b128(unsigned lds_off, const void* gaddr) {
  asm volatile("global_load_async_to_lds_b128 %0, %1, off"
               :: "v"(lds_off), "v"((unsigned long long)(size_t)gaddr)
               : "memory");
}
__device__ __forceinline__ void wait_async0() {
  asm volatile("s_wait_asynccnt 0" ::: "memory");
}

// --------------------------- prep kernel -----------------------------------
struct PrepP {
  const float *c1w, *c2w, *c3w, *bn1g, *bn2g, *bn3g;
  const float *pw, *pb, *A, *Bm, *Cm, *ow;
  const float *g1w, *g2w, *h1w, *h2w;
  float    *bu;
  unsigned *c1Pk, *c2Pk, *c3Pk, *rnnPk, *wlcPk, *g1Pk, *g2Pk, *h1Pk, *h2Pk;
};

__global__ __launch_bounds__(256) void prep_kernel(PrepP p) {
  const int tid = threadIdx.x;
  const float rs = rsqrtf(1.f + 1e-5f);

  if (tid < DST) {                       // b_u = proj_b @ Bm
    float s = 0.f;
    for (int h = 0; h < H_DIM; ++h) s += p.pb[h] * p.Bm[h * DST + tid];
    p.bu[tid] = s;
  }
  // conv1 pack: NT=4, CH=1  (K = k*4+i, K<12)
  for (int w = tid; w < 4 * 256; w += 256) {
    int lane = w & 31, r = (w >> 5) & 7, nt = w >> 8;
    int g = lane >> 4, n = lane & 15, o = nt * 16 + n, k0 = ((r >> 2) << 4) + (g << 3) + ((r & 3) << 1);
    float vv[2];
    for (int q = 0; q < 2; ++q) {
      int K = k0 + q; float val = 0.f;
      if (K < 12) { int i = K & 3, k = K >> 2; val = p.c1w[o * 12 + i * 3 + k] * p.bn1g[o] * rs; }
      vv[q] = val;
    }
    p.c1Pk[w] = pkh(vv[0], vv[1]);
  }
  // conv2 pack: NT=4, CH=10 (K = k*64+i, K<320)
  for (int w = tid; w < 4 * 10 * 256; w += 256) {
    int lane = w & 31, r = (w >> 5) & 7, u = w >> 8, nt = u / 10, c = u % 10;
    int g = lane >> 4, n = lane & 15, o = nt * 16 + n, k0 = c * 32 + g * 16 + 2 * r;
    float vv[2];
    for (int q = 0; q < 2; ++q) {
      int K = k0 + q; float val = 0.f;
      if (K < 320) { int i = K & 63, k = K >> 6; val = p.c2w[o * 320 + i * 5 + k] * p.bn2g[o] * rs; }
      vv[q] = val;
    }
    p.c2Pk[w] = pkh(vv[0], vv[1]);
  }
  // conv3 pack: NT=4, CH=14 (K<448)
  for (int w = tid; w < 4 * 14 * 256; w += 256) {
    int lane = w & 31, r = (w >> 5) & 7, u = w >> 8, nt = u / 14, c = u % 14;
    int g = lane >> 4, n = lane & 15, o = nt * 16 + n, k0 = c * 32 + g * 16 + 2 * r;
    float vv[2];
    for (int q = 0; q < 2; ++q) {
      int K = k0 + q; float val = 0.f;
      if (K < 448) { int i = K & 63, k = K >> 6; val = p.c3w[o * 448 + i * 7 + k] * p.bn3g[o] * rs; }
      vv[q] = val;
    }
    p.c3Pk[w] = pkh(vv[0], vv[1]);
  }
  // rnn pack: rows 0..3 = W_u = proj_w@Bm ; rows 4..19 = A^T ; rest 0
  for (int w = tid; w < 256; w += 256) {
    int lane = w & 31, r = (w >> 5) & 7, g = lane >> 4, n = lane & 15;
    int k0 = ((r >> 2) << 4) + (g << 3) + ((r & 3) << 1);
    float vv[2];
    for (int q = 0; q < 2; ++q) {
      int K = k0 + q; float val = 0.f;
      if (K < 4) {
        float s = 0.f;
        for (int h = 0; h < H_DIM; ++h) s += p.pw[K * H_DIM + h] * p.Bm[h * DST + n];
        val = s;
      } else if (K < 20) {
        val = p.A[n * DST + (K - 4)];      // B[k][n] = A^T[k-4][n] = A[n][k-4]
      }
      vv[q] = val;
    }
    p.rnnPk[w] = pkh(vv[0], vv[1]);
  }
  // wlc pack: Wlc = Cm @ out_w (16x64), rows >=16 zero
  for (int w = tid; w < 4 * 256; w += 256) {
    int lane = w & 31, r = (w >> 5) & 7, nt = w >> 8;
    int g = lane >> 4, n = lane & 15, j = nt * 16 + n, k0 = ((r >> 2) << 4) + (g << 3) + ((r & 3) << 1);
    float vv[2];
    for (int q = 0; q < 2; ++q) {
      int K = k0 + q; float val = 0.f;
      if (K < 16) {
        float s = 0.f;
        for (int h = 0; h < H_DIM; ++h) s += p.Cm[K * H_DIM + h] * p.ow[h * H_DIM + j];
        val = s;
      }
      vv[q] = val;
    }
    p.wlcPk[w] = pkh(vv[0], vv[1]);
  }
  // g1 pack: (128,64)  NT=4 CH=4
  for (int w = tid; w < 16 * 256; w += 256) {
    int lane = w & 31, r = (w >> 5) & 7, u = w >> 8, nt = u >> 2, c = u & 3;
    int g = lane >> 4, n = lane & 15, o = nt * 16 + n, k0 = c * 32 + g * 16 + 2 * r;
    p.g1Pk[w] = pkh(p.g1w[k0 * H_DIM + o], p.g1w[(k0 + 1) * H_DIM + o]);
  }
  // g2 pack: (64,64)  NT=4 CH=2
  for (int w = tid; w < 8 * 256; w += 256) {
    int lane = w & 31, r = (w >> 5) & 7, u = w >> 8, nt = u >> 1, c = u & 1;
    int g = lane >> 4, n = lane & 15, o = nt * 16 + n, k0 = c * 32 + g * 16 + 2 * r;
    p.g2Pk[w] = pkh(p.g2w[k0 * H_DIM + o], p.g2w[(k0 + 1) * H_DIM + o]);
  }
  // h1 pack: (64,32)  NT=2 CH=2
  for (int w = tid; w < 4 * 256; w += 256) {
    int lane = w & 31, r = (w >> 5) & 7, u = w >> 8, nt = u >> 1, c = u & 1;
    int g = lane >> 4, n = lane & 15, j = nt * 16 + n, k0 = c * 32 + g * 16 + 2 * r;
    p.h1Pk[w] = pkh(p.h1w[k0 * 32 + j], p.h1w[(k0 + 1) * 32 + j]);
  }
  // h2 pack: (32,3)   NT=1 CH=1, cols >=3 zero
  for (int w = tid; w < 256; w += 256) {
    int lane = w & 31, r = (w >> 5) & 7, g = lane >> 4, n = lane & 15;
    int k0 = g * 16 + 2 * r;
    float v0 = (n < 3) ? p.h2w[k0 * 3 + n] : 0.f;
    float v1 = (n < 3) ? p.h2w[(k0 + 1) * 3 + n] : 0.f;
    p.h2Pk[w] = pkh(v0, v1);
  }
}

// --------------------------- RNN kernel ------------------------------------
// One wave = 16 batch rows. Per step: one WMMA  [x_t | h](16x32) @ Brnn(32x16).
// x tile async-copied to LDS (raw f32); h round-trips through a 512B LDS tile
// for the C-layout -> A-layout transpose (LDS ops are in-order per wave).
struct RnnP {
  const float *x, *outb, *bu;
  const unsigned *rnnPk, *wlcPk;
  float *lc;
};

__global__ __launch_bounds__(64) void rnn_kernel(RnnP p) {
  const int lane = threadIdx.x & 31;
  const int wave = threadIdx.x >> 5;
  const int g = lane >> 4, n = lane & 15, m = n;
  const int base = (blockIdx.x * 2 + wave) * 16;

  __shared__ __align__(16) float     xs[2][16][SLONG][4];  // raw f32 x tile
  __shared__ __align__(16) _Float16  hb[2][16][DST];       // h (wave-private)

  // async-stage x tile (16 rows x 100 t x 4 ch), 16B per async op
  for (int idx = lane; idx < 16 * SLONG; idx += 32) {
    int mm = idx / SLONG, t = idx % SLONG;
    async_load_b128((unsigned)(size_t)&xs[wave][mm][t][0],
                    p.x + ((size_t)(base + mm) * SLONG + t) * 4);
  }
  for (int idx = lane; idx < 16 * DST / 2; idx += 32)
    ((unsigned*)&hb[wave][0][0])[idx] = 0u;   // h0 = 0

  AOp bop;
  for (int r = 0; r < 8; ++r) bop.u[r] = p.rnnPk[r * 32 + lane];
  const float bias = p.bu[n];

  wait_async0();

  for (int t = 0; t < SLONG; ++t) {
    // branchless A-operand: whole h row (32B) as two b128 LDS loads + selects
    float4 xv = *(const float4*)&xs[wave][m][t][0];
    unsigned x01 = pkh(xv.x, xv.y), x23 = pkh(xv.z, xv.w);
    uint4 hlo = *(const uint4*)&hb[wave][m][0];   // s0..7  pairs
    uint4 hhi = *(const uint4*)&hb[wave][m][8];   // s8..15 pairs
    AOp a;
    a.u[0] = g ? hlo.z : x01;   // K8,9  = s4,5   | K0,1  = x0,1
    a.u[1] = g ? hlo.w : x23;   // K10,11= s6,7   | K2,3  = x2,3
    a.u[2] = g ? hhi.x : hlo.x; // K12,13= s8,9   | K4,5  = s0,1
    a.u[3] = g ? hhi.y : hlo.y; // K14,15= s10,11 | K6,7  = s2,3
    a.u[4] = g ? 0u    : hhi.z; // K24.. = 0      | K16,17= s12,13
    a.u[5] = g ? 0u    : hhi.w; // K26.. = 0      | K18,19= s14,15
    a.u[6] = 0u; a.u[7] = 0u;
    v8f c; for (int r = 0; r < 8; ++r) c[r] = bias;
    c = wmma_f16(a, bop, c);
    for (int r = 0; r < 8; ++r)
      hb[wave][r + 8 * g][n] = (_Float16)tanh_f(c[r]);
  }

  // long_context = h_final @ Wlc + out_b  (K=16 padded to 32)
  {
    uint4 hlo = *(const uint4*)&hb[wave][m][0];
    uint4 hhi = *(const uint4*)&hb[wave][m][8];
    AOp ha;
    ha.u[0] = g ? hhi.x : hlo.x;
    ha.u[1] = g ? hhi.y : hlo.y;
    ha.u[2] = g ? hhi.z : hlo.z;
    ha.u[3] = g ? hhi.w : hlo.w;
    ha.u[4] = 0u; ha.u[5] = 0u; ha.u[6] = 0u; ha.u[7] = 0u;
    for (int nt = 0; nt < 4; ++nt) {
      AOp bw;
      for (int r = 0; r < 8; ++r) bw.u[r] = p.wlcPk[(nt * 8 + r) * 32 + lane];
      float ob = p.outb[nt * 16 + n];
      v8f c; for (int r = 0; r < 8; ++r) c[r] = ob;
      c = wmma_f16(ha, bw, c);
      for (int r = 0; r < 8; ++r)
        p.lc[(size_t)(base + r + 8 * g) * H_DIM + nt * 16 + n] = c[r];
    }
  }
}

// --------------------- short conv path + head kernel ------------------------
struct HeadP {
  const float *x;
  const float *c1b, *c2b, *c3b, *bn1g, *bn1b, *bn2g, *bn2b, *bn3g, *bn3b;
  const float *g1b, *g2b, *h1b, *h2b;
  const unsigned *c1Pk, *c2Pk, *c3Pk, *g1Pk, *g2Pk, *h1Pk, *h2Pk;
  const float *lc;
  float *out;
};

union Region1 {
  _Float16 a1[16][14][64];                                   // conv1 out (pad 2)
  struct {
    _Float16 cb[16][128]; _Float16 tb[16][64];
    _Float16 fb[16][64];  _Float16 rb[16][32];
  } hd;                                                      // head scratch
};

__global__ __launch_bounds__(32) void short_head_kernel(HeadP p) {
  const int lane = threadIdx.x & 31, g = lane >> 4, n = lane & 15, m = n;
  const int base = blockIdx.x * 16;

  __shared__ __align__(16) unsigned  xsw[16][12][2];   // x_short packed (pad 1)
  __shared__ __align__(16) Region1   r1;
  __shared__ __align__(16) _Float16  a2[16][16][64];   // conv2 out (pad 3)

  __builtin_prefetch(p.lc + (size_t)base * H_DIM, 0, 0);

  // async-stage raw x_short into (not-yet-used) a2 storage, then pack to f16
  {
    char* stage = (char*)&a2[0][0][0];     // 2560B scratch inside a2
    for (int idx = lane; idx < 160; idx += 32) {
      int mm = idx / 10, t = idx % 10;
      async_load_b128((unsigned)(size_t)(stage + idx * 16),
                      p.x + ((size_t)(base + mm) * SLONG + 90 + t) * 4);
    }
    wait_async0();
    for (int idx = lane; idx < 160; idx += 32) {
      int mm = idx / 10, t = idx % 10;
      float4 v = *(const float4*)(stage + idx * 16);
      xsw[mm][t + 1][0] = pkh(v.x, v.y);
      xsw[mm][t + 1][1] = pkh(v.z, v.w);
    }
  }
  // zero pad slots only (a2 pads after staging scratch is consumed)
  for (int idx = lane; idx < 64; idx += 32) {
    int mm = idx >> 2, s = (idx >> 1) & 1, w = idx & 1;
    xsw[mm][s ? 11 : 0][w] = 0u;
  }
  {
    const int pt1[4] = {0, 1, 12, 13};
    for (int idx = lane; idx < 16 * 4 * 32; idx += 32) {
      int mm = idx >> 7, tt = pt1[(idx >> 5) & 3], w = idx & 31;
      ((unsigned*)&r1.a1[mm][tt][0])[w] = 0u;
    }
    const int pt2[6] = {0, 1, 2, 13, 14, 15};
    for (int idx = lane; idx < 16 * 6 * 32; idx += 32) {
      int mm = idx / 192, tt = pt2[(idx >> 5) % 6], w = idx & 31;
      ((unsigned*)&a2[mm][tt][0])[w] = 0u;
    }
  }

  const float rs = rsqrtf(1.f + 1e-5f);
  float b1f[4], b2f[4], b3f[4];
  for (int nt = 0; nt < 4; ++nt) {
    int o = nt * 16 + n;
    b1f[nt] = p.bn1g[o] * rs * p.c1b[o] + p.bn1b[o];
    b2f[nt] = p.bn2g[o] * rs * p.c2b[o] + p.bn2b[o];
    b3f[nt] = p.bn3g[o] * rs * p.c3b[o] + p.bn3b[o];
  }

  // conv1: M=16 batch, N=64 (4 tiles), K=12 (pad 32) -- branchless A
  for (int t = 0; t < SSHORT; ++t) {
    unsigned w00 = xsw[m][t][0],     w01 = xsw[m][t][1];
    unsigned w10 = xsw[m][t + 1][0], w11 = xsw[m][t + 1][1];
    unsigned w20 = xsw[m][t + 2][0], w21 = xsw[m][t + 2][1];
    AOp a;
    a.u[0] = g ? w20 : w00;
    a.u[1] = g ? w21 : w01;
    a.u[2] = g ? 0u : w10;
    a.u[3] = g ? 0u : w11;
    a.u[4] = 0u; a.u[5] = 0u; a.u[6] = 0u; a.u[7] = 0u;
    for (int nt = 0; nt < 4; ++nt) {
      AOp bw;
      for (int r = 0; r < 8; ++r) bw.u[r] = p.c1Pk[(nt * 8 + r) * 32 + lane];
      v8f c; for (int r = 0; r < 8; ++r) c[r] = b1f[nt];
      c = wmma_f16(a, bw, c);
      for (int r = 0; r < 8; ++r)
        r1.a1[r + 8 * g][t + 2][nt * 16 + n] = (_Float16)fmaxf(c[r], 0.f);
    }
  }
  // conv2: K = 64*5 = 320 -> 10 chunks
  for (int t = 0; t < SSHORT; ++t) {
    v8f acc[4];
    for (int nt = 0; nt < 4; ++nt) for (int r = 0; r < 8; ++r) acc[nt][r] = b2f[nt];
    for (int c = 0; c < 10; ++c) {
      int k = c >> 1, ib = (c & 1) * 32;
      AOp a;
      for (int r = 0; r < 8; ++r) a.u[r] = ldpair(&r1.a1[m][t + k][ib + kpA(r, g)]);
      for (int nt = 0; nt < 4; ++nt) {
        AOp bw;
        for (int r = 0; r < 8; ++r) bw.u[r] = p.c2Pk[((nt * 10 + c) * 8 + r) * 32 + lane];
        acc[nt] = wmma_f16(a, bw, acc[nt]);
      }
    }
    for (int nt = 0; nt < 4; ++nt) for (int r = 0; r < 8; ++r)
      a2[r + 8 * g][t + 3][nt * 16 + n] = (_Float16)fmaxf(acc[nt][r], 0.f);
  }
  // conv3 (K = 64*7 = 448 -> 14 chunks) + time-mean
  v8f macc[4];
  for (int nt = 0; nt < 4; ++nt) for (int r = 0; r < 8; ++r) macc[nt][r] = 0.f;
  for (int t = 0; t < SSHORT; ++t) {
    v8f acc[4];
    for (int nt = 0; nt < 4; ++nt) for (int r = 0; r < 8; ++r) acc[nt][r] = b3f[nt];
    for (int c = 0; c < 14; ++c) {
      int k = c >> 1, ib = (c & 1) * 32;
      AOp a;
      for (int r = 0; r < 8; ++r) a.u[r] = ldpair(&a2[m][t + k][ib + kpA(r, g)]);
      for (int nt = 0; nt < 4; ++nt) {
        AOp bw;
        for (int r = 0; r < 8; ++r) bw.u[r] = p.c3Pk[((nt * 14 + c) * 8 + r) * 32 + lane];
        acc[nt] = wmma_f16(a, bw, acc[nt]);
      }
    }
    for (int nt = 0; nt < 4; ++nt) for (int r = 0; r < 8; ++r)
      macc[nt][r] += fmaxf(acc[nt][r], 0.f);
  }
  for (int nt = 0; nt < 4; ++nt) for (int r = 0; r < 8; ++r) macc[nt][r] *= 0.1f;

  // combined = [short | long]  (16 x 128)
  for (int nt = 0; nt < 4; ++nt) for (int r = 0; r < 8; ++r)
    r1.hd.cb[r + 8 * g][nt * 16 + n] = (_Float16)macc[nt][r];
  for (int idx = lane; idx < 16 * H_DIM; idx += 32) {
    int mm = idx >> 6, j = idx & 63;
    r1.hd.cb[mm][64 + j] = (_Float16)p.lc[(size_t)(base + mm) * H_DIM + j];
  }
  // g1: (16x128)@(128x64) -> tanh
  for (int nt = 0; nt < 4; ++nt) {
    v8f c; float bb = p.g1b[nt * 16 + n];
    for (int r = 0; r < 8; ++r) c[r] = bb;
    for (int cc = 0; cc < 4; ++cc) {
      AOp a;
      for (int r = 0; r < 8; ++r) a.u[r] = ldpair(&r1.hd.cb[m][cc * 32 + kpA(r, g)]);
      AOp bw;
      for (int r = 0; r < 8; ++r) bw.u[r] = p.g1Pk[((nt * 4 + cc) * 8 + r) * 32 + lane];
      c = wmma_f16(a, bw, c);
    }
    for (int r = 0; r < 8; ++r)
      r1.hd.tb[r + 8 * g][nt * 16 + n] = (_Float16)tanh_f(c[r]);
  }
  // g2: sigmoid -> gate; filtered = short * gate
  for (int nt = 0; nt < 4; ++nt) {
    v8f c; float bb = p.g2b[nt * 16 + n];
    for (int r = 0; r < 8; ++r) c[r] = bb;
    for (int cc = 0; cc < 2; ++cc) {
      AOp a;
      for (int r = 0; r < 8; ++r) a.u[r] = ldpair(&r1.hd.tb[m][cc * 32 + kpA(r, g)]);
      AOp bw;
      for (int r = 0; r < 8; ++r) bw.u[r] = p.g2Pk[((nt * 2 + cc) * 8 + r) * 32 + lane];
      c = wmma_f16(a, bw, c);
    }
    for (int r = 0; r < 8; ++r) {
      float f = macc[nt][r] * sigm_f(c[r]);
      r1.hd.fb[r + 8 * g][nt * 16 + n] = (_Float16)f;
    }
  }
  // h1: (16x64)@(64x32) -> relu
  for (int nt = 0; nt < 2; ++nt) {
    v8f c; float bb = p.h1b[nt * 16 + n];
    for (int r = 0; r < 8; ++r) c[r] = bb;
    for (int cc = 0; cc < 2; ++cc) {
      AOp a;
      for (int r = 0; r < 8; ++r) a.u[r] = ldpair(&r1.hd.fb[m][cc * 32 + kpA(r, g)]);
      AOp bw;
      for (int r = 0; r < 8; ++r) bw.u[r] = p.h1Pk[((nt * 2 + cc) * 8 + r) * 32 + lane];
      c = wmma_f16(a, bw, c);
    }
    for (int r = 0; r < 8; ++r)
      r1.hd.rb[r + 8 * g][nt * 16 + n] = (_Float16)fmaxf(c[r], 0.f);
  }
  // h2: (16x32)@(32x3) -> logits
  {
    v8f c; float bb = (n < 3) ? p.h2b[n] : 0.f;
    for (int r = 0; r < 8; ++r) c[r] = bb;
    AOp a;
    for (int r = 0; r < 8; ++r) a.u[r] = ldpair(&r1.hd.rb[m][kpA(r, g)]);
    AOp bw;
    for (int r = 0; r < 8; ++r) bw.u[r] = p.h2Pk[r * 32 + lane];
    c = wmma_f16(a, bw, c);
    if (n < 3) {
      for (int r = 0; r < 8; ++r)
        p.out[(size_t)(base + r + 8 * g) * 3 + n] = c[r];
    }
  }
}

// ------------------------------ launcher ------------------------------------
extern "C" void kernel_launch(void* const* d_in, const int* in_sizes, int n_in,
                              void* d_out, int out_size, void* d_ws, size_t ws_size,
                              hipStream_t stream) {
  const float* x      = (const float*)d_in[0];
  const float* c1w    = (const float*)d_in[1];
  const float* c1b    = (const float*)d_in[2];
  const float* c2w    = (const float*)d_in[3];
  const float* c2b    = (const float*)d_in[4];
  const float* c3w    = (const float*)d_in[5];
  const float* c3b    = (const float*)d_in[6];
  const float* bn1g   = (const float*)d_in[7];
  const float* bn1b   = (const float*)d_in[8];
  const float* bn2g   = (const float*)d_in[9];
  const float* bn2b   = (const float*)d_in[10];
  const float* bn3g   = (const float*)d_in[11];
  const float* bn3b   = (const float*)d_in[12];
  const float* pw     = (const float*)d_in[13];
  const float* pb     = (const float*)d_in[14];
  const float* A      = (const float*)d_in[15];
  const float* Bm     = (const float*)d_in[16];
  const float* Cm     = (const float*)d_in[17];
  const float* ow     = (const float*)d_in[18];
  const float* outb   = (const float*)d_in[19];
  const float* g1w    = (const float*)d_in[20];
  const float* g1b    = (const float*)d_in[21];
  const float* g2w    = (const float*)d_in[22];
  const float* g2b    = (const float*)d_in[23];
  const float* h1w    = (const float*)d_in[24];
  const float* h1b    = (const float*)d_in[25];
  const float* h2w    = (const float*)d_in[26];
  const float* h2b    = (const float*)d_in[27];

  const int Btot = in_sizes[0] / (SLONG * 4);      // 16384

  char* ws = (char*)d_ws;
  float*    lc  = (float*)ws;
  size_t    lcB = (size_t)Btot * H_DIM * sizeof(float);
  float*    bu  = (float*)(ws + lcB);
  unsigned* pk  = (unsigned*)(ws + lcB + 64);
  unsigned* c1Pk  = pk;
  unsigned* c2Pk  = pk + 1024;
  unsigned* c3Pk  = pk + 11264;
  unsigned* rnnPk = pk + 25600;
  unsigned* wlcPk = pk + 25856;
  unsigned* g1Pk  = pk + 26880;
  unsigned* g2Pk  = pk + 30976;
  unsigned* h1Pk  = pk + 33024;
  unsigned* h2Pk  = pk + 34048;

  PrepP pp{c1w, c2w, c3w, bn1g, bn2g, bn3g,
           pw, pb, A, Bm, Cm, ow,
           g1w, g2w, h1w, h2w,
           bu, c1Pk, c2Pk, c3Pk, rnnPk, wlcPk, g1Pk, g2Pk, h1Pk, h2Pk};
  prep_kernel<<<dim3(1), dim3(256), 0, stream>>>(pp);

  RnnP rp{x, outb, bu, rnnPk, wlcPk, lc};
  rnn_kernel<<<dim3(Btot / 32), dim3(64), 0, stream>>>(rp);

  HeadP hp{x, c1b, c2b, c3b, bn1g, bn1b, bn2g, bn2b, bn3g, bn3b,
           g1b, g2b, h1b, h2b,
           c1Pk, c2Pk, c3Pk, g1Pk, g2Pk, h1Pk, h2Pk,
           lc, (float*)d_out};
  short_head_kernel<<<dim3(Btot / 16), dim3(32), 0, stream>>>(hp);
}
